// PAC_32830730010712
// MI455X (gfx1250) — matrix operand order repaired and checked by reading, not compiled, not run on hardware
//
#include <hip/hip_runtime.h>
#include <hip/hip_bf16.h>

typedef __attribute__((ext_vector_type(16))) _Float16 v16h;
typedef __attribute__((ext_vector_type(8)))  float    v8f;

#define FSAMP     512.0f
#define T_LEN     4096
#define N_SIG     64          // B*C = 4*16
#define N_PHA     10
#define N_AMP     10
#define FIR_LEN   513
#define KCHUNKS   17          // 17*32 = 544 >= 513 taps
#define K_PAD     (KCHUNKS * 32)
#define N_BINS    18
#define TCHUNK    512
#define PI_F      3.14159265358979f
#define TWO_PI_F  6.28318530717959f

// Analytic filterbank, stored pre-packed in WMMA B-matrix lane order:
//   [bank(re/im)][ntile(2)][kc(17)][lane(32)][e(16)]  as f16
// Element (lane,e) of a 32x16 f16 B tile holds B[K = (e>>3)*16 + (lane>>4)*8 + (e&7), N = lane&15].
#define HF_ELEMS  (2 * 2 * KCHUNKS * 32 * 16)   // 34816 halves = 69632 B

// ---------------------------------------------------------------------------
// Kernel 1: build the analytic (complex) FIR bank in WMMA-ready layout.
// bank 0 = given real bandpass taps; bank 1 = analytic quadrature pair
//   g(t) = (cos(w_lo t) - cos(w_hi t)) / (pi t) * hamming  (g(0) = 0)
// ---------------------------------------------------------------------------
__global__ void pac_build_filters(const float* __restrict__ filters,
                                  _Float16* __restrict__ hf) {
    int gid = blockIdx.x * blockDim.x + threadIdx.x;
    if (gid >= HF_ELEMS) return;
    int e     = gid & 15;
    int lane  = (gid >> 4) & 31;
    int kc    = (gid >> 9) % KCHUNKS;
    int rest  = gid / (16 * 32 * KCHUNKS);
    int ntile = rest & 1;
    int bank  = rest >> 1;

    int band = ntile * 16 + (lane & 15);
    int k    = kc * 32 + (e >> 3) * 16 + (lane >> 4) * 8 + (e & 7);

    float val = 0.0f;
    if (band < 20 && k < FIR_LEN) {
        if (bank == 0) {
            val = filters[band * FIR_LEN + k];
        } else {
            float c = (band < N_PHA) ? (2.0f + 2.0f * (float)band)
                                     : (60.0f + (100.0f / 9.0f) * (float)(band - N_PHA));
            float t = (float)(k - (FIR_LEN - 1) / 2);
            if (t != 0.0f) {
                float wlo = TWO_PI_F * (0.75f * c) / FSAMP;
                float whi = TWO_PI_F * (1.25f * c) / FSAMP;
                float ham = 0.54f - 0.46f * __cosf(TWO_PI_F * (float)k / (float)(FIR_LEN - 1));
                val = (__cosf(wlo * t) - __cosf(whi * t)) / (PI_F * t) * ham;
            }
        }
    }
    hf[gid] = (_Float16)val;
}

// ---------------------------------------------------------------------------
// Kernel 2: analytic bandpass via WMMA GEMM.
//   Y[t, n] = sum_k x[t - 256 + k] * Hf[k, n],  n in {re,im} x {band tiles}
// Block = 256 thr (8 waves). Block covers 512 time samples of one signal;
// each wave owns 4 16-row M tiles. x staged in LDS with halo (ds traffic);
// B tiles are single aligned 32B loads from the pre-packed bank (L2 resident).
// Epilogue converts accumulators to phase (atan2) / amplitude (hypot).
// ---------------------------------------------------------------------------
__global__ void __launch_bounds__(256) pac_filter_hilbert(
    const float* __restrict__ x, const _Float16* __restrict__ hf,
    float* __restrict__ phase, float* __restrict__ amp) {
    __shared__ float xs[TCHUNK + K_PAD];   // 1056 floats: [chunkbase-256, ...)

    const int sig = blockIdx.y;
    const int chunkbase = blockIdx.x * TCHUNK;
    const int tid = threadIdx.x;

    __builtin_prefetch(hf, 0, 3);          // global_prefetch of filter bank

    const float* xsig = x + sig * T_LEN;
    for (int j = tid; j < TCHUNK + K_PAD; j += 256) {
        int pos = chunkbase - (FIR_LEN - 1) / 2 + j;
        xs[j] = (pos >= 0 && pos < T_LEN) ? xsig[pos] : 0.0f;
    }
    __syncthreads();

    const int wave = tid >> 5, lane = tid & 31;
    const int m = lane & 15, hs = lane >> 4;
    const v16h* hfv = (const v16h*)hf;

    for (int i = 0; i < 4; ++i) {
        const int t0loc = (wave * 4 + i) * 16;
        v8f aR0 = {}, aR1 = {}, aI0 = {}, aI1 = {};
        for (int kc = 0; kc < KCHUNKS; ++kc) {
            // A tile (16x32 f16): element e -> K = (e>>3)*16 + hs*8 + (e&7), M = m
            v16h a;
            const int abase = t0loc + m + kc * 32 + hs * 8;
            #pragma unroll
            for (int e = 0; e < 16; ++e)
                a[e] = (_Float16)xs[abase + (e >> 3) * 16 + (e & 7)];
            // B tiles: one v16h per lane, pre-packed
            v16h bR0 = hfv[((0 * 2 + 0) * KCHUNKS + kc) * 32 + lane];
            v16h bR1 = hfv[((0 * 2 + 1) * KCHUNKS + kc) * 32 + lane];
            v16h bI0 = hfv[((1 * 2 + 0) * KCHUNKS + kc) * 32 + lane];
            v16h bI1 = hfv[((1 * 2 + 1) * KCHUNKS + kc) * 32 + lane];
            aR0 = __builtin_amdgcn_wmma_f32_16x16x32_f16(false, a, false, bR0, (short)0, aR0, false, false);
            aR1 = __builtin_amdgcn_wmma_f32_16x16x32_f16(false, a, false, bR1, (short)0, aR1, false, false);
            aI0 = __builtin_amdgcn_wmma_f32_16x16x32_f16(false, a, false, bI0, (short)0, aI0, false, false);
            aI1 = __builtin_amdgcn_wmma_f32_16x16x32_f16(false, a, false, bI1, (short)0, aI1, false, false);
        }
        // D element (lane, r): time = t0 + r + 8*hs, band = (lane&15) [+16]
        #pragma unroll
        for (int r = 0; r < 8; ++r) {
            int tglob = chunkbase + t0loc + r + 8 * hs;
            int band0 = m;                 // ntile 0: bands 0..15
            if (band0 < N_PHA) {
                phase[(sig * N_PHA + band0) * T_LEN + tglob] = atan2f(aI0[r], aR0[r]);
            } else {
                amp[(sig * N_AMP + (band0 - N_PHA)) * T_LEN + tglob] =
                    sqrtf(aR0[r] * aR0[r] + aI0[r] * aI0[r]);
            }
            int band1 = 16 + m;            // ntile 1: bands 16..19 valid
            if (band1 < 20) {
                amp[(sig * N_AMP + (band1 - N_PHA)) * T_LEN + tglob] =
                    sqrtf(aR1[r] * aR1[r] + aI1[r] * aI1[r]);
            }
        }
    }
}

// ---------------------------------------------------------------------------
// Kernel 3: soft phase binning + Tort MI. One block per (signal, phase band).
// amp_sum[k, a] and cnt[k] computed as a single WMMA GEMM:
//   A (bins x time, 2 M-tiles for 18 bins) x B (time x [10 amps | ones | 0])
// Each of 8 waves owns disjoint 32-sample K chunks; f32 partials merged with
// ds_add_f32, then 10 threads finish the MI.
// ---------------------------------------------------------------------------
__global__ void __launch_bounds__(256) pac_mi(
    const float* __restrict__ phase, const float* __restrict__ amp,
    float* __restrict__ out) {
    __shared__ float ph[T_LEN];
    __shared__ float red[32 * 16];

    const int sig = blockIdx.x / N_PHA;
    const int p   = blockIdx.x % N_PHA;
    const int tid = threadIdx.x;

    const float* psrc = phase + (sig * N_PHA + p) * T_LEN;
    for (int j = tid; j < T_LEN; j += 256) ph[j] = psrc[j];
    for (int j = tid; j < 32 * 16; j += 256) red[j] = 0.0f;
    __syncthreads();

    const int wave = tid >> 5, lane = tid & 31;
    const int m = lane & 15, hs = lane >> 4;
    const float step = TWO_PI_F / (float)N_BINS;
    const float* asig = amp + sig * N_AMP * T_LEN;

    v8f acc0 = {}, acc1 = {};
    for (int c = wave; c < T_LEN / 32; c += 8) {
        const int t0 = c * 32;
        v16h a0, a1, bm;
        #pragma unroll
        for (int e = 0; e < 16; ++e) {
            int tt = t0 + (e >> 3) * 16 + hs * 8 + (e & 7);
            float phv = ph[tt];
            // softmax over 18 bins at this sample; keep this lane's bins
            float s = 0.0f, n0 = 0.0f, n1 = 0.0f;
            for (int k = 0; k < N_BINS; ++k) {
                float ck = -PI_F + ((float)k + 0.5f) * step;
                float d = phv - ck + PI_F;
                d -= TWO_PI_F * floorf(d * (1.0f / TWO_PI_F));
                d -= PI_F;
                float ek = __expf(-d * d * 100.0f);   // 1/TEMP = 100
                s += ek;
                n0 = (k == m) ? ek : n0;
                n1 = (k == 16 + m) ? ek : n1;
            }
            float inv = 1.0f / s;
            a0[e] = (_Float16)(n0 * inv);
            a1[e] = (m < 2) ? (_Float16)(n1 * inv) : (_Float16)0.0f;
            // B (time x N): cols 0..9 = amplitudes, col 10 = 1 (gives cnt)
            int tb = t0 + (e >> 3) * 16 + hs * 8 + (e & 7);
            float bv = (m < N_AMP) ? asig[m * T_LEN + tb]
                                   : ((m == N_AMP) ? 1.0f : 0.0f);
            bm[e] = (_Float16)bv;
        }
        acc0 = __builtin_amdgcn_wmma_f32_16x16x32_f16(false, a0, false, bm, (short)0, acc0, false, false);
        acc1 = __builtin_amdgcn_wmma_f32_16x16x32_f16(false, a1, false, bm, (short)0, acc1, false, false);
    }
    // cross-wave reduction: red[row=bin][col=n]; tile1 rows land at 16+r == bin
    #pragma unroll
    for (int r = 0; r < 8; ++r) {
        int row = r + 8 * hs;
        atomicAdd(&red[row * 16 + m], acc0[r]);
        atomicAdd(&red[(16 + row) * 16 + m], acc1[r]);
    }
    __syncthreads();

    if (tid < N_AMP) {
        float ms[N_BINS], tot = 0.0f;
        for (int k = 0; k < N_BINS; ++k) {
            float v = red[k * 16 + tid] / (red[k * 16 + N_AMP] + 1e-8f);
            ms[k] = v; tot += v;
        }
        const float logK = 2.8903717578961645f;   // log(18)
        float acc = logK;
        for (int k = 0; k < N_BINS; ++k) {
            float pk = ms[k] / (tot + 1e-8f);
            acc += pk * __logf(pk + 1e-8f);
        }
        out[blockIdx.x * N_AMP + tid] = acc / logK;
    }
}

// ---------------------------------------------------------------------------
extern "C" void kernel_launch(void* const* d_in, const int* in_sizes, int n_in,
                              void* d_out, int out_size, void* d_ws, size_t ws_size,
                              hipStream_t stream) {
    (void)in_sizes; (void)n_in; (void)out_size; (void)ws_size;
    const float* x       = (const float*)d_in[0];   // (4,16,4096) f32
    const float* filters = (const float*)d_in[1];   // (20,513)    f32
    float* out = (float*)d_out;                     // (4,16,10,10) f32

    char* ws = (char*)d_ws;
    _Float16* hf  = (_Float16*)ws;                            // 69,632 B
    float* phase  = (float*)(ws + 69632);                     // 10,485,760 B
    float* ampbuf = (float*)(ws + 69632 + 10485760);          // 10,485,760 B

    pac_build_filters<<<(HF_ELEMS + 255) / 256, 256, 0, stream>>>(filters, hf);
    pac_filter_hilbert<<<dim3(T_LEN / TCHUNK, N_SIG), 256, 0, stream>>>(x, hf, phase, ampbuf);
    pac_mi<<<N_SIG * N_PHA, 256, 0, stream>>>(phase, ampbuf, out);
}
